// GraphSAGE_32916629356559
// MI455X (gfx1250) — matrix operand order repaired and checked by reading, not compiled, not run on hardware
//
#include <hip/hip_runtime.h>
#include <stdint.h>

#define N_NODES  100000
#define N_EDGES  1600000
#define FDIM     128
#define BN_EPS   1e-5f
#define MT_TILES 6250                 // N_NODES / 16 (exact)
#define MT_GROUPS 1563                // ceil(6250 / 4)
#define ROW_PAD  (MT_GROUPS * 64)     // 100032 padded rows for bf16 operands

// ---- CDNA5 WMMA vector types (wave32) ----
typedef __bf16 bf16x16 __attribute__((ext_vector_type(16)));
typedef float  f32x8   __attribute__((ext_vector_type(8)));
typedef int    i32x4   __attribute__((ext_vector_type(4)));

// address-space-qualified int4 pointers for the async builtin
typedef __attribute__((address_space(1))) i32x4 gas_i32x4;   // global
typedef __attribute__((address_space(3))) i32x4 las_i32x4;   // LDS

union FragBF {           // 16 bf16 = 32 bytes = two b128 chunks
    bf16x16 m;
    uint4   q[2];
};

// float -> bf16, round-to-nearest-even
static __device__ __forceinline__ unsigned short f2bf(float f) {
    unsigned int u = __float_as_uint(f);
    unsigned int r = u + 0x7FFFu + ((u >> 16) & 1u);
    return (unsigned short)(r >> 16);
}

// ---- async global->LDS 16B copy (ASYNCcnt) with safe sync fallback ----
#if __has_builtin(__builtin_amdgcn_global_load_async_to_lds_b128)
#define SAGE_HAS_ASYNC 1
#else
#define SAGE_HAS_ASYNC 0
#endif

static __device__ __forceinline__ void async_copy16(const void* g, void* l) {
#if SAGE_HAS_ASYNC
    __builtin_amdgcn_global_load_async_to_lds_b128(
        (gas_i32x4*)g, (las_i32x4*)l, 0, 0);
#else
    *(uint4*)l = *(const uint4*)g;    // vmem load + ds_store_b128
#endif
}

static __device__ __forceinline__ void async_wait0() {
#if SAGE_HAS_ASYNC
#if __has_builtin(__builtin_amdgcn_s_wait_asynccnt)
    __builtin_amdgcn_s_wait_asynccnt(0);
#else
    asm volatile("s_wait_asynccnt 0x0" ::: "memory");
#endif
#endif
}

// -------------------- zero fill --------------------
__global__ void sage_zero_f32(float* __restrict__ p, int n) {
    int i = blockIdx.x * blockDim.x + threadIdx.x;
    if (i < n) p[i] = 0.0f;
}

// ------------- edge scatter: agg[dst] += x[src]; deg[dst] += 1 -------------
// one wave (32 lanes) per edge, 4 floats per lane => 128 features
__global__ __launch_bounds__(256) void sage_scatter(
    const float* __restrict__ x,
    const long long* __restrict__ src,
    const long long* __restrict__ dst,
    float* __restrict__ agg,
    float* __restrict__ deg)
{
    int gid  = blockIdx.x * blockDim.x + threadIdx.x;
    int e    = gid >> 5;
    int lane = gid & 31;
    if (e >= N_EDGES) return;
    long long s = src[e];
    long long d = dst[e];
    const float4 v = *(const float4*)(x + (size_t)s * FDIM + lane * 4);
    float* base = agg + (size_t)d * FDIM + lane * 4;
    atomicAdd(base + 0, v.x);
    atomicAdd(base + 1, v.y);
    atomicAdd(base + 2, v.z);
    atomicAdd(base + 3, v.w);
    if (lane == 0) atomicAdd(deg + d, 1.0f);
}

// ---- convert activations to bf16: xbf = bf16(h); abf = bf16(agg / max(deg,1)) ----
__global__ void sage_convert(
    const float* __restrict__ h, const float* __restrict__ agg,
    const float* __restrict__ deg,
    unsigned short* __restrict__ xbf, unsigned short* __restrict__ abf)
{
    int i = blockIdx.x * blockDim.x + threadIdx.x;
    if (i >= N_NODES * FDIM) return;
    int row = i >> 7;
    float dg = deg[row];
    dg = dg > 1.0f ? dg : 1.0f;
    xbf[i] = f2bf(h[i]);
    abf[i] = f2bf(agg[i] / dg);
}

// ---- pack W (K x Nout, row-major f32) into WMMA B-fragment order (bf16) ----
// fragment f = nt*Ktiles + kt : 32 lanes x 16 bf16 contiguous
// lane holds column n = nt*16 + (lane&15), K = kt*32 + (lane>=16 ? 16:0) + j
__global__ void sage_pack_w(const float* __restrict__ W,
                            unsigned short* __restrict__ out, int Nout)
{
    int idx = blockIdx.x * blockDim.x + threadIdx.x;
    if (idx >= FDIM * Nout) return;
    int j = idx & 15;
    int l = (idx >> 4) & 31;
    int f = idx >> 9;
    const int Ktiles = FDIM / 32;          // 4
    int kt = f % Ktiles;
    int nt = f / Ktiles;
    int k = kt * 32 + ((l >> 4) << 4) + j;
    int n = nt * 16 + (l & 15);
    out[idx] = f2bf(W[(size_t)k * Nout + n]);
}

// ---- Z = A_bf @ WlP + X_bf @ WrP + b   via v_wmma_f32_16x16x32_bf16 ----
// block = 4 waves; block covers 64 rows x 64 cols.
// Weights for the block's 4 column-tiles staged in LDS (async), then each
// wave computes a 64x16 strip: 4 M-tiles reuse each B fragment (register
// blocking) -> 32 WMMAs per wave.
__global__ __launch_bounds__(128) void sage_gemm_wmma(
    const unsigned short* __restrict__ Abf,
    const unsigned short* __restrict__ Xbf,
    const unsigned short* __restrict__ WlP,
    const unsigned short* __restrict__ WrP,
    const float* __restrict__ bias,
    float* __restrict__ Z,
    int Nout)
{
    __shared__ __align__(16) unsigned short lWl[8192];   // 4 nt x 4 kt x 512 bf16
    __shared__ __align__(16) unsigned short lWr[8192];

    const int ntGroups = Nout >> 6;                 // 2 (dout=128) or 1 (dout=64)
    int mg = blockIdx.x / ntGroups;
    int ng = blockIdx.x - mg * ntGroups;
    int ntBase = ng * 4;

    // ---- stage packed weights for 4 column-tiles into LDS ----
    int tid = threadIdx.x;
    const unsigned short* wlsrc = WlP + (size_t)ntBase * 2048;   // 4*Ktiles*512
    const unsigned short* wrsrc = WrP + (size_t)ntBase * 2048;
#pragma unroll
    for (int i = 0; i < 4; ++i) {                   // 512 x 16B chunks / 128 thr
        int c = tid + i * 128;
        async_copy16(wlsrc + c * 8, &lWl[c * 8]);
        async_copy16(wrsrc + c * 8, &lWr[c * 8]);
    }
    async_wait0();
    __syncthreads();

    int wave = tid >> 5;
    int lane = tid & 31;
    int nt = ntBase + wave;                         // this wave's column tile
    int m0 = mg * 64;                               // 4 row tiles m0..m0+63

    // A-fragment lane geometry (16-bit A 16x32, contiguous k, same order as packed B):
    // lane<16: row=lane, K 0..15; lane>=16: row=lane-16, K 16..31
    int mrow  = m0 + (lane & 15);
    int khalf = (lane >> 4) << 4;

    const unsigned short* arow = Abf + (size_t)mrow * FDIM;
    const unsigned short* xrow = Xbf + (size_t)mrow * FDIM;
    __builtin_prefetch(arow, 0, 0);                 // global_prefetch_b8
    __builtin_prefetch(xrow, 0, 0);

    f32x8 acc[4] = {};
#pragma unroll
    for (int kt = 0; kt < 4; ++kt) {
        int kbase = kt * 32 + khalf;
        // B fragments from LDS (shared across the 4 M-tiles below)
        FragBF fbL, fbR;
        const uint4* bl = (const uint4*)&lWl[((wave * 4 + kt) * 32 + lane) * 16];
        const uint4* br = (const uint4*)&lWr[((wave * 4 + kt) * 32 + lane) * 16];
        fbL.q[0] = bl[0]; fbL.q[1] = bl[1];
        fbR.q[0] = br[0]; fbR.q[1] = br[1];
#pragma unroll
        for (int j = 0; j < 4; ++j) {
            const unsigned short* aj = arow + (size_t)(j * 16) * FDIM + kbase;
            const unsigned short* xj = xrow + (size_t)(j * 16) * FDIM + kbase;
            FragBF fa;
            fa.q[0] = *(const uint4*)(aj);
            fa.q[1] = *(const uint4*)(aj + 8);
            acc[j] = __builtin_amdgcn_wmma_f32_16x16x32_bf16(
                false, fa.m, false, fbL.m, (short)0, acc[j], false, false);
            fa.q[0] = *(const uint4*)(xj);
            fa.q[1] = *(const uint4*)(xj + 8);
            acc[j] = __builtin_amdgcn_wmma_f32_16x16x32_bf16(
                false, fa.m, false, fbR.m, (short)0, acc[j], false, false);
        }
    }

    // C/D layout: VGPR r, lanes 0-15 -> M=r, lanes 16-31 -> M=r+8; N = lane&15
    int ncol = nt * 16 + (lane & 15);
    float bval = bias[ncol];
    int rsub = (lane >> 4) << 3;
#pragma unroll
    for (int j = 0; j < 4; ++j) {
        int tileRow = m0 + j * 16;
        if (tileRow < N_NODES) {                    // wave-uniform tail guard
            float* zp = Z + (size_t)(tileRow + rsub) * Nout + ncol;
#pragma unroll
            for (int r = 0; r < 8; ++r)
                zp[(size_t)r * Nout] = acc[j][r] + bval;
        }
    }
}

// ---- BN column stats: stats[c] += sum, stats[Nout+c] += sumsq ----
__global__ void sage_bn_stats(const float* __restrict__ Z,
                              float* __restrict__ stats, int Nout)
{
    int col = threadIdx.x;                 // blockDim == Nout
    int r0 = blockIdx.x * 512;
    int r1 = r0 + 512;
    if (r1 > N_NODES) r1 = N_NODES;
    float s1 = 0.0f, s2 = 0.0f;
    for (int r = r0; r < r1; ++r) {
        float v = Z[(size_t)r * Nout + col];
        s1 += v;
        s2 += v * v;
    }
    atomicAdd(&stats[col], s1);
    atomicAdd(&stats[Nout + col], s2);
}

// ---- BN apply (+ optional ReLU) ----
__global__ void sage_bn_apply(const float* __restrict__ Z,
                              const float* __restrict__ stats,
                              const float* __restrict__ gamma,
                              const float* __restrict__ beta,
                              float* __restrict__ out, int Nout, int relu)
{
    int i = blockIdx.x * blockDim.x + threadIdx.x;
    if (i >= N_NODES * Nout) return;
    int col = i % Nout;
    const float invN = 1.0f / (float)N_NODES;
    float mean = stats[col] * invN;
    float var  = stats[Nout + col] * invN - mean * mean;
    float y = gamma[col] * (Z[i] - mean) * rsqrtf(var + BN_EPS) + beta[col];
    if (relu) y = fmaxf(y, 0.0f);
    out[i] = y;
}

extern "C" void kernel_launch(void* const* d_in, const int* in_sizes, int n_in,
                              void* d_out, int out_size, void* d_ws, size_t ws_size,
                              hipStream_t stream) {
    (void)in_sizes; (void)n_in; (void)out_size; (void)ws_size;

    const float*     x    = (const float*)d_in[0];
    const long long* ei   = (const long long*)d_in[1];   // int64 edge_index (2, E)
    const long long* srcp = ei;
    const long long* dstp = ei + N_EDGES;

    // ---- workspace bump allocator (256B aligned) ----
    char*  ws  = (char*)d_ws;
    size_t off = 0;
    auto alloc = [&](size_t bytes) -> char* {
        char* p = ws + off;
        off += (bytes + 255) & ~(size_t)255;
        return p;
    };
    float*          agg   = (float*)alloc((size_t)N_NODES * FDIM * 4);  // also GEMM output Z
    float*          h0    = (float*)alloc((size_t)N_NODES * FDIM * 4);
    float*          h1    = (float*)alloc((size_t)N_NODES * FDIM * 4);
    float*          deg   = (float*)alloc((size_t)N_NODES * 4);
    unsigned short* abf   = (unsigned short*)alloc((size_t)ROW_PAD * FDIM * 2);
    unsigned short* xbf   = (unsigned short*)alloc((size_t)ROW_PAD * FDIM * 2);
    unsigned short* wlP   = (unsigned short*)alloc((size_t)FDIM * FDIM * 2);
    unsigned short* wrP   = (unsigned short*)alloc((size_t)FDIM * FDIM * 2);
    float*          stats = (float*)alloc(2 * FDIM * 4);

    const int douts[3] = {128, 128, 64};
    const float* hin = x;

    for (int i = 0; i < 3; ++i) {
        const int dout = douts[i];
        const float* Wl = (const float*)d_in[2 + 5 * i];
        const float* Wr = (const float*)d_in[3 + 5 * i];
        const float* b  = (const float*)d_in[4 + 5 * i];
        const float* g  = (const float*)d_in[5 + 5 * i];
        const float* bt = (const float*)d_in[6 + 5 * i];
        float* hout = (i == 2) ? (float*)d_out : (i == 0 ? h0 : h1);

        // zero accumulators
        sage_zero_f32<<<(N_NODES * FDIM + 255) / 256, 256, 0, stream>>>(agg, N_NODES * FDIM);
        sage_zero_f32<<<(N_NODES + 255) / 256, 256, 0, stream>>>(deg, N_NODES);
        sage_zero_f32<<<1, 256, 0, stream>>>(stats, 2 * dout);

        // mean aggregation (scatter) — one wave per edge
        sage_scatter<<<(N_EDGES * 32) / 256, 256, 0, stream>>>(hin, srcp, dstp, agg, deg);

        // bf16 conversion of both GEMM A operands
        sage_convert<<<(N_NODES * FDIM + 255) / 256, 256, 0, stream>>>(hin, agg, deg, xbf, abf);

        // pack weights into WMMA B-fragment order
        sage_pack_w<<<(FDIM * dout + 255) / 256, 256, 0, stream>>>(Wl, wlP, dout);
        sage_pack_w<<<(FDIM * dout + 255) / 256, 256, 0, stream>>>(Wr, wrP, dout);

        // fused dual GEMM + bias  (Z reuses agg buffer); block = 64 rows x 64 cols
        const int blocks = MT_GROUPS * (dout >> 6);
        sage_gemm_wmma<<<blocks, 128, 0, stream>>>(abf, xbf, wlP, wrP, b, agg, dout);

        // batchnorm (training-mode, biased var) + ReLU on hidden layers
        sage_bn_stats<<<(N_NODES + 511) / 512, dout, 0, stream>>>(agg, stats, dout);
        sage_bn_apply<<<(N_NODES * dout + 255) / 256, 256, 0, stream>>>(agg, stats, g, bt,
                                                                        hout, dout, i < 2 ? 1 : 0);
        hin = hout;
    }
}
